// SovarielOmega_90074054132489
// MI455X (gfx1250) — matrix operand order repaired and checked by reading, not compiled, not run on hardware
//
#include <hip/hip_runtime.h>
#include <hip/hip_bf16.h>
#include <math.h>

// ---------------------------------------------------------------------------
// MI455X (gfx1250): f16-storage dataflow, all GEMMs via v_wmma_f32_16x16x32_f16
// with pure vector-load fragments (no per-iteration f32->f16 conversion).
// Shapes: B=2048, S=8, tokens=16384, E=512, H=256, NH=8.
// ---------------------------------------------------------------------------

typedef __attribute__((ext_vector_type(16))) _Float16 v16h;
typedef __attribute__((ext_vector_type(8)))  _Float16 v8h;
typedef __attribute__((ext_vector_type(4)))  _Float16 v4h;
typedef __attribute__((ext_vector_type(8)))  float    v8f;

#define B_   2048
#define S_   8
#define TOK_ (B_ * S_)
#define E_   512
#define H_   256

__device__ __forceinline__ float sigm_(float x) { return 1.f / (1.f + __expf(-x)); }

__device__ __forceinline__ v16h cat8h_(v8h lo, v8h hi) {
    return __builtin_shufflevector(lo, hi, 0,1,2,3,4,5,6,7,8,9,10,11,12,13,14,15);
}

// ---------------------------------------------------------------------------
// Bulk f32 -> f16 conversion (weights once per launch, raw inputs).
// ---------------------------------------------------------------------------
__global__ __launch_bounds__(256) void cvt_f32_f16_kernel(
    const float* __restrict__ in, _Float16* __restrict__ out, int n)
{
    const int i = (blockIdx.x * 256 + threadIdx.x) * 4;
    if (i + 3 < n) {
        const float4 v = *(const float4*)(in + i);
        v4h h = { (_Float16)v.x, (_Float16)v.y, (_Float16)v.z, (_Float16)v.w };
        *(v4h*)(out + i) = h;
    }
}

// ---------------------------------------------------------------------------
// GEMM: C = act(A[MxK]f16 * (W[NxK]f16)^T + bias0 + bias1)
// Output to f32 (Cf) and/or f16 (Ch) buffers.
// Block 256 threads = 8 waves; wave tile 32 rows x 64 cols (2x4 wmma accs,
// B fragment reused across both row fragments). Block tile 64 x 256.
// Requires M%64==0, N%256==0, K%32==0.
// Fragment layouts per CDNA5 ISA:
//   A 16x32 f16: lanes 0-15 row M=lane hold K {0..7,16..23}; lanes 16-31
//                hold K {8..15,24..31}  -> two contiguous 16B loads.
//   B 32x16 f16: lane n=(lane&15), same K split as A -> two contiguous 16B loads
//                (W row-major gives each 8-wide K run of W[n] contiguously).
// ---------------------------------------------------------------------------
__global__ __launch_bounds__(256) void gemm_wmma_f16_kernel(
    const _Float16* __restrict__ A, int lda,
    const _Float16* __restrict__ W, int ldw,
    const float* __restrict__ bias0, const float* __restrict__ bias1,
    float* __restrict__ Cf, _Float16* __restrict__ Ch,
    int ldc, int col_off, int K, int do_relu)
{
    const int lane = threadIdx.x & 31;
    const int wv   = threadIdx.x >> 5;
    const int row0 = blockIdx.y * 64  + (wv & 1) * 32;
    const int n0   = blockIdx.x * 256 + (wv >> 1) * 64;

    const v8f vz = {0.f,0.f,0.f,0.f,0.f,0.f,0.f,0.f};
    v8f acc[2][4] = {{vz,vz,vz,vz},{vz,vz,vz,vz}};

    const int half = lane >> 4;
    const int r15  = lane & 15;

    const _Float16* a0 = A + (size_t)(row0 + r15) * lda + half * 8;
    const _Float16* a1 = a0 + (size_t)16 * lda;
    const _Float16* wb = W + (size_t)(n0 + r15) * ldw + half * 8;

    for (int kk = 0; kk < K; kk += 32) {
        if (kk + 32 < K) {
            __builtin_prefetch(a0 + kk + 32, 0, 0);   // global_prefetch_b8
            __builtin_prefetch(a1 + kk + 32, 0, 0);
        }
        const v16h af0 = cat8h_(*(const v8h*)(a0 + kk), *(const v8h*)(a0 + kk + 16));
        const v16h af1 = cat8h_(*(const v8h*)(a1 + kk), *(const v8h*)(a1 + kk + 16));

        #pragma unroll
        for (int tn = 0; tn < 4; ++tn) {
            const v16h b = cat8h_(*(const v8h*)(wb + (size_t)(tn * 16) * ldw + kk), *(const v8h*)(wb + (size_t)(tn * 16) * ldw + kk + 16));
            acc[0][tn] = __builtin_amdgcn_wmma_f32_16x16x32_f16(
                false, af0, false, b, (short)0, acc[0][tn], false, false);
            acc[1][tn] = __builtin_amdgcn_wmma_f32_16x16x32_f16(
                false, af1, false, b, (short)0, acc[1][tn], false, false);
        }
    }

    // C layout: lane l holds col (l&15); vgpr r -> row r + 8*(l>>4).
    const int rbase = half * 8;
    #pragma unroll
    for (int i = 0; i < 2; ++i) {
        #pragma unroll
        for (int tn = 0; tn < 4; ++tn) {
            const int n = n0 + tn * 16 + r15;
            float bv = 0.f;
            if (bias0) bv += bias0[n];
            if (bias1) bv += bias1[n];
            #pragma unroll
            for (int r = 0; r < 8; ++r) {
                float val = acc[i][tn][r] + bv;
                if (do_relu) val = fmaxf(val, 0.f);
                const size_t idx = (size_t)(row0 + i * 16 + rbase + r) * ldc + col_off + n;
                if (Cf) Cf[idx] = val;
                if (Ch) Ch[idx] = (_Float16)val;
            }
        }
    }
}

// ---------------------------------------------------------------------------
// One LSTM timestep: G = P[:,t] + h_in * Whh^T (f16 WMMA); gate nonlinearity;
// update c (f32), write h_out (f16, ping-ponged) and layer-output slice (f16).
// Grid = (H/64, B/16), block = 128 (4 waves, one 16x16 hidden tile each,
// 4 gate accumulators per wave).
// ---------------------------------------------------------------------------
__global__ __launch_bounds__(128) void lstm_step_kernel(
    const float* __restrict__ P,        // [TOK x 4H] pre-gates (biases folded)
    const _Float16* __restrict__ Whh,   // [4H x H] f16
    const _Float16* __restrict__ h_in,  // [B x H] f16
    _Float16* __restrict__ h_out,       // [B x H] f16
    float* __restrict__ c,              // [B x H] f32 (in/out)
    _Float16* __restrict__ out,         // [TOK x 2H] f16, cols [col_off, +H)
    int t_actual, int col_off)
{
    const int lane = threadIdx.x & 31;
    const int wv   = threadIdx.x >> 5;
    const int j0   = blockIdx.x * 64 + wv * 16;
    const int rb0  = blockIdx.y * 16;

    const v8f vz = {0.f,0.f,0.f,0.f,0.f,0.f,0.f,0.f};
    v8f acc[4] = {vz, vz, vz, vz};      // i, f, g, o

    const int half = lane >> 4;
    const int r15  = lane & 15;

    const _Float16* ap = h_in + (size_t)(rb0 + r15) * H_ + half * 8;
    const _Float16* wb = Whh + (size_t)(j0 + r15) * H_ + half * 8;

    #pragma unroll
    for (int kk = 0; kk < H_; kk += 32) {
        const v16h a = cat8h_(*(const v8h*)(ap + kk), *(const v8h*)(ap + kk + 16));
        #pragma unroll
        for (int g = 0; g < 4; ++g) {
            const v16h b = cat8h_(*(const v8h*)(wb + (size_t)(g * H_) * H_ + kk), *(const v8h*)(wb + (size_t)(g * H_) * H_ + kk + 16));
            acc[g] = __builtin_amdgcn_wmma_f32_16x16x32_f16(
                false, a, false, b, (short)0, acc[g], false, false);
        }
    }

    const int j     = j0 + r15;
    const int rbase = half * 8;
    #pragma unroll
    for (int r = 0; r < 8; ++r) {
        const int rb = rb0 + rbase + r;
        const size_t tok = (size_t)rb * S_ + t_actual;
        const float* prow = P + tok * (4 * H_);
        const float gi = acc[0][r] + prow[0 * H_ + j];
        const float gf = acc[1][r] + prow[1 * H_ + j];
        const float gg = acc[2][r] + prow[2 * H_ + j];
        const float go = acc[3][r] + prow[3 * H_ + j];

        const size_t sidx = (size_t)rb * H_ + j;
        const float cv = sigm_(gf) * c[sidx] + sigm_(gi) * tanhf(gg);
        const float hv = sigm_(go) * tanhf(cv);
        c[sidx]     = cv;
        h_out[sidx] = (_Float16)hv;
        out[tok * (2 * H_) + col_off + j] = (_Float16)hv;
    }
}

// ---------------------------------------------------------------------------
// Attention (S=8, NH=8, hd=64): one block per batch; FLOP-trivial VALU path.
// Writes attention output as f16 (feeds the Wo GEMM) + head-mean attn weights.
// ---------------------------------------------------------------------------
__global__ __launch_bounds__(256) void attn_kernel(
    const float* __restrict__ qkv,       // [TOK x 1536] (q | k | v) f32
    _Float16* __restrict__ attn_pre,     // [TOK x 512] f16
    float* __restrict__ attn_w)          // [B x 8 x 8]
{
    const int b   = blockIdx.x;
    const int tid = threadIdx.x;
    __shared__ float sc[64];
    __shared__ float accw[64];

    if (tid < 64) accw[tid] = 0.f;
    __syncthreads();

    for (int hh = 0; hh < 8; ++hh) {
        if (tid < 64) {
            const int q = tid >> 3, k = tid & 7;
            const float* qp = qkv + ((size_t)b * S_ + q) * 1536 + hh * 64;
            const float* kp = qkv + ((size_t)b * S_ + k) * 1536 + 512 + hh * 64;
            float s = 0.f;
            #pragma unroll 8
            for (int d = 0; d < 64; ++d) s += qp[d] * kp[d];
            sc[tid] = s * 0.125f;        // 1/sqrt(64)
        }
        __syncthreads();
        if (tid < 8) {                   // softmax over row tid
            float m = -1e30f;
            for (int k = 0; k < 8; ++k) m = fmaxf(m, sc[tid * 8 + k]);
            float sum = 0.f;
            for (int k = 0; k < 8; ++k) { float e = __expf(sc[tid * 8 + k] - m); sc[tid * 8 + k] = e; sum += e; }
            const float inv = 1.f / sum;
            for (int k = 0; k < 8; ++k) sc[tid * 8 + k] *= inv;
        }
        __syncthreads();
        if (tid < 64) accw[tid] += sc[tid] * 0.125f;   // mean over heads
        for (int i = tid; i < 512; i += 256) {
            const int q = i >> 6, e = i & 63;
            float o = 0.f;
            #pragma unroll
            for (int k = 0; k < 8; ++k)
                o += sc[q * 8 + k] * qkv[((size_t)b * S_ + k) * 1536 + 1024 + hh * 64 + e];
            attn_pre[((size_t)b * S_ + q) * E_ + hh * 64 + e] = (_Float16)o;
        }
        __syncthreads();
    }
    if (tid < 64) attn_w[(size_t)b * 64 + tid] = accw[tid];
}

// ---------------------------------------------------------------------------
// Head: weights = softmax(mean_q attn_w); summary = sum_s w[s]*attn_out[b,s];
// coherence = min(sigmoid(relu(summary@Wc1^T+bc1)@Wc2^T+bc2), 0.9998).
// ---------------------------------------------------------------------------
__global__ __launch_bounds__(256) void head_kernel(
    const float* __restrict__ attn_out,  // [TOK x 512] f32
    const float* __restrict__ attn_w,    // [B x 8 x 8]
    const float* __restrict__ Wc1,       // [256 x 513]
    const float* __restrict__ bc1,
    const float* __restrict__ Wc2,       // [1 x 256]
    const float* __restrict__ bc2,
    float* __restrict__ out)             // [B]
{
    const int b   = blockIdx.x;
    const int tid = threadIdx.x;
    __shared__ float wts[8];
    __shared__ float summary[512];
    __shared__ float red[256];

    if (tid < 8) {
        float s = 0.f;
        for (int q = 0; q < 8; ++q) s += attn_w[(size_t)b * 64 + q * 8 + tid];
        wts[tid] = s * 0.125f;
    }
    __syncthreads();
    if (tid == 0) {
        float m = -1e30f;
        for (int k = 0; k < 8; ++k) m = fmaxf(m, wts[k]);
        float sum = 0.f;
        for (int k = 0; k < 8; ++k) { float e = __expf(wts[k] - m); wts[k] = e; sum += e; }
        const float inv = 1.f / sum;
        for (int k = 0; k < 8; ++k) wts[k] *= inv;
    }
    __syncthreads();
    for (int e = tid; e < 512; e += 256) {
        float s = 0.f;
        #pragma unroll
        for (int q = 0; q < 8; ++q) s += attn_out[((size_t)b * S_ + q) * E_ + e] * wts[q];
        summary[e] = s;
    }
    __syncthreads();
    float hsum = bc1[tid];
    const float* wrow = Wc1 + (size_t)tid * 513;     // col 512 multiplies zero
    #pragma unroll 8
    for (int e = 0; e < 512; ++e) hsum += wrow[e] * summary[e];
    red[tid] = fmaxf(hsum, 0.f) * Wc2[tid];
    __syncthreads();
    for (int s = 128; s > 0; s >>= 1) {
        if (tid < s) red[tid] += red[tid + s];
        __syncthreads();
    }
    if (tid == 0) {
        const float z = red[0] + bc2[0];
        out[b] = fminf(1.f / (1.f + __expf(-z)), 0.9998f);
    }
}

// ---------------------------------------------------------------------------
// Host orchestration
// ---------------------------------------------------------------------------
static inline void launch_cvt(const float* in, _Float16* out, size_t n, hipStream_t s)
{
    const int blocks = (int)((n + 1023) / 1024);
    cvt_f32_f16_kernel<<<blocks, 256, 0, s>>>(in, out, (int)n);
}

static inline void launch_gemm(const _Float16* A, int lda, const _Float16* W, int ldw,
                               const float* b0, const float* b1,
                               float* Cf, _Float16* Ch, int ldc, int col_off,
                               int M, int N, int K, int relu, hipStream_t s)
{
    dim3 grid(N / 256, M / 64), blk(256);
    gemm_wmma_f16_kernel<<<grid, blk, 0, s>>>(A, lda, W, ldw, b0, b1, Cf, Ch,
                                              ldc, col_off, K, relu);
}

extern "C" void kernel_launch(void* const* d_in, const int* in_sizes, int n_in,
                              void* d_out, int out_size, void* d_ws, size_t ws_size,
                              hipStream_t stream)
{
    (void)in_sizes; (void)n_in; (void)out_size; (void)ws_size;

    const float* audio = (const float*)d_in[0];
    const float* visual= (const float*)d_in[1];
    const float* Wa1 = (const float*)d_in[2];  const float* ba1 = (const float*)d_in[3];
    const float* Wa2 = (const float*)d_in[4];  const float* ba2 = (const float*)d_in[5];
    const float* Wv1 = (const float*)d_in[6];  const float* bv1 = (const float*)d_in[7];
    const float* Wv2 = (const float*)d_in[8];  const float* bv2 = (const float*)d_in[9];
    const float* Wih = (const float*)d_in[10]; const float* Whh = (const float*)d_in[11];
    const float* bih = (const float*)d_in[12]; const float* bhh = (const float*)d_in[13];
    const float* Wqkv= (const float*)d_in[14]; const float* bqkv= (const float*)d_in[15];
    const float* Wo  = (const float*)d_in[16]; const float* bo  = (const float*)d_in[17];
    const float* Wc1 = (const float*)d_in[18]; const float* bc1 = (const float*)d_in[19];
    const float* Wc2 = (const float*)d_in[20]; const float* bc2 = (const float*)d_in[21];

    // ---- workspace carve-up (256B aligned bump allocator) ------------------
    char* base = (char*)d_ws;
    size_t cur = 0;
    auto carve = [&](size_t bytes) -> char* {
        char* p = base + cur;
        cur += (bytes + 255) & ~(size_t)255;
        return p;
    };
    _Float16* raw16  = (_Float16*)carve((size_t)TOK_ * 1024 * 2);  // audio/visual staging (reused)
    _Float16* tmp16  = (_Float16*)carve((size_t)TOK_ * 512 * 2);   // front MLP hidden
    _Float16* seq16  = (_Float16*)carve((size_t)TOK_ * 512 * 2);
    _Float16* l1_16  = (_Float16*)carve((size_t)TOK_ * 512 * 2);
    _Float16* l2_16  = (_Float16*)carve((size_t)TOK_ * 512 * 2);
    float*    P      = (float*)   carve((size_t)TOK_ * 1024 * 4);  // pre-gates
    _Float16* h16a   = (_Float16*)carve((size_t)B_ * H_ * 2);
    _Float16* h16b   = (_Float16*)carve((size_t)B_ * H_ * 2);
    float*    cst    = (float*)   carve((size_t)B_ * H_ * 4);
    float*    qkv    = (float*)   carve((size_t)TOK_ * 1536 * 4);
    _Float16* apre16 = (_Float16*)carve((size_t)TOK_ * 512 * 2);
    float*    aw     = (float*)   carve((size_t)B_ * 64 * 4);
    float*    aout   = (float*)   carve((size_t)TOK_ * 512 * 4);
    _Float16* Wa1h   = (_Float16*)carve((size_t)512 * 1024 * 2);
    _Float16* Wa2h   = (_Float16*)carve((size_t)256 * 512 * 2);
    _Float16* Wv1h   = (_Float16*)carve((size_t)512 * 1024 * 2);
    _Float16* Wv2h   = (_Float16*)carve((size_t)256 * 512 * 2);
    _Float16* Wihh   = (_Float16*)carve((size_t)4 * 1024 * 512 * 2);
    _Float16* Whhh   = (_Float16*)carve((size_t)4 * 1024 * 256 * 2);
    _Float16* Wqkvh  = (_Float16*)carve((size_t)1536 * 512 * 2);
    _Float16* Woh    = (_Float16*)carve((size_t)512 * 512 * 2);

    // ---- one-time weight conversion to f16 --------------------------------
    launch_cvt(Wa1,  Wa1h,  (size_t)512 * 1024,      stream);
    launch_cvt(Wa2,  Wa2h,  (size_t)256 * 512,       stream);
    launch_cvt(Wv1,  Wv1h,  (size_t)512 * 1024,      stream);
    launch_cvt(Wv2,  Wv2h,  (size_t)256 * 512,       stream);
    launch_cvt(Wih,  Wihh,  (size_t)4 * 1024 * 512,  stream);
    launch_cvt(Whh,  Whhh,  (size_t)4 * 1024 * 256,  stream);
    launch_cvt(Wqkv, Wqkvh, (size_t)1536 * 512,      stream);
    launch_cvt(Wo,   Woh,   (size_t)512 * 512,       stream);

    // ---- Front-end MLPs: audio -> seq[:, :256], visual -> seq[:, 256:512]
    launch_cvt(audio, raw16, (size_t)TOK_ * 1024, stream);
    launch_gemm(raw16, 1024, Wa1h, 1024, ba1, nullptr, nullptr, tmp16, 512, 0,
                TOK_, 512, 1024, 1, stream);
    launch_gemm(tmp16, 512, Wa2h, 512, ba2, nullptr, nullptr, seq16, 512, 0,
                TOK_, 256, 512, 0, stream);
    launch_cvt(visual, raw16, (size_t)TOK_ * 1024, stream);
    launch_gemm(raw16, 1024, Wv1h, 1024, bv1, nullptr, nullptr, tmp16, 512, 0,
                TOK_, 512, 1024, 1, stream);
    launch_gemm(tmp16, 512, Wv2h, 512, bv2, nullptr, nullptr, seq16, 512, 256,
                TOK_, 256, 512, 0, stream);

    // ---- 2-layer bidirectional LSTM ---------------------------------------
    const _Float16* layer_in  = seq16;
    _Float16*       layer_out = l1_16;
    for (int l = 0; l < 2; ++l) {
        for (int d = 0; d < 2; ++d) {
            const int ld = l * 2 + d;
            const _Float16* Wih_ld = Wihh + (size_t)ld * (4 * H_) * 512;
            const _Float16* Whh_ld = Whhh + (size_t)ld * (4 * H_) * H_;
            const float* bih_ld = bih + (size_t)ld * (4 * H_);
            const float* bhh_ld = bhh + (size_t)ld * (4 * H_);

            // Pre-gates for all tokens (bih + bhh folded in as dual bias).
            launch_gemm(layer_in, 512, Wih_ld, 512, bih_ld, bhh_ld,
                        P, nullptr, 4 * H_, 0, TOK_, 4 * H_, 512, 0, stream);

            hipMemsetAsync(h16a, 0, (size_t)B_ * H_ * sizeof(_Float16), stream);
            hipMemsetAsync(cst,  0, (size_t)B_ * H_ * sizeof(float), stream);

            _Float16* hin = h16a; _Float16* hout = h16b;
            for (int t = 0; t < S_; ++t) {
                const int ta = d ? (S_ - 1 - t) : t;
                lstm_step_kernel<<<dim3(H_ / 64, B_ / 16), 128, 0, stream>>>(
                    P, Whh_ld, hin, hout, cst, layer_out, ta, d * H_);
                _Float16* tmp = hin; hin = hout; hout = tmp;
            }
        }
        layer_in  = layer_out;
        layer_out = l2_16;
    }
    const _Float16* lstm_out = l2_16;

    // ---- Multi-head attention ---------------------------------------------
    launch_gemm(lstm_out, 512, Wqkvh, 512, bqkv, nullptr, qkv, nullptr, 1536, 0,
                TOK_, 1536, 512, 0, stream);
    attn_kernel<<<B_, 256, 0, stream>>>(qkv, apre16, aw);
    launch_gemm(apre16, 512, Woh, 512, bo, nullptr, aout, nullptr, 512, 0,
                TOK_, 512, 512, 0, stream);

    // ---- Summary + coherence head -----------------------------------------
    head_kernel<<<B_, 256, 0, stream>>>(aout, aw, Wc1, bc1, Wc2, bc2, (float*)d_out);
}
